// SPC_SA_19370302505069
// MI455X (gfx1250) — matrix operand order repaired and checked by reading, not compiled, not run on hardware
//
#include <hip/hip_runtime.h>

#define NB 16
#define CCH 768
#define NTEMB 512
#define NHID 1536
#define KSEL 384
#define HW 4096

typedef __attribute__((ext_vector_type(16))) __bf16 bf16x16;
typedef __attribute__((ext_vector_type(8))) float floatx8;

#if defined(__has_builtin)
#if __has_builtin(__builtin_amdgcn_global_load_async_to_lds_b128) && \
    __has_builtin(__builtin_amdgcn_s_wait_asynccnt)
#define SPC_ASYNC 1
#endif
#endif
#ifndef SPC_ASYNC
#define SPC_ASYNC 0
#endif

// async-to-LDS builtin operand types (from hipcc diagnostic):
//   param0: int4-vector pointer in global (AS1 / __device__) space
//   param1: int4-vector pointer in shared (AS3) space
typedef int v4i __attribute__((__vector_size__(4 * sizeof(int))));
typedef __attribute__((address_space(1))) v4i gas_v4i;
typedef __attribute__((address_space(3))) v4i lds_v4i;

__device__ __forceinline__ unsigned short f32_to_bf16(float f) {
  unsigned int u = __float_as_uint(f);
  u += 0x7FFFu + ((u >> 16) & 1u);
  return (unsigned short)(u >> 16);
}
__device__ __forceinline__ float silu_f(float x) { return x / (1.0f + expf(-x)); }

// ---------------------------------------------------------------------------
// mean over pixels of x: xmean[b][c]
__global__ __launch_bounds__(256) void spc_xmean(const float* __restrict__ x,
                                                 float* __restrict__ xmean) {
  __shared__ float red[256];
  int b = blockIdx.y, c = blockIdx.x, t = threadIdx.x;
  const float* xp = x + ((long)(b * CCH + c)) * HW;
  float s = 0.0f;
  for (int i = t; i < HW; i += 256) s += xp[i];
  red[t] = s; __syncthreads();
  for (int st = 128; st > 0; st >>= 1) {
    if (t < st) red[t] += red[t + st];
    __syncthreads();
  }
  if (t == 0) xmean[b * CCH + c] = red[0] * (1.0f / (float)HW);
}

// silu(temb) -> st
__global__ void spc_silu_temb(const float* __restrict__ temb, float* __restrict__ st) {
  int i = blockIdx.x * 256 + threadIdx.x;
  if (i < NB * NTEMB) st[i] = silu_f(temb[i]);
}

// ss[b][0..1535] = st@W_ts^T + b_ts ; tp[b][0..767] = st@W_tp^T + b_tp
__global__ void spc_time_proj(const float* __restrict__ st,
                              const float* __restrict__ W_ts, const float* __restrict__ b_ts,
                              const float* __restrict__ W_tp, const float* __restrict__ b_tp,
                              float* __restrict__ ss, float* __restrict__ tp) {
  int i = blockIdx.x * 256 + threadIdx.x;
  if (i >= NB * (2 * CCH + CCH)) return;
  int b = i / (3 * CCH);
  int col = i % (3 * CCH);
  const float* sb = st + b * NTEMB;
  if (col < 2 * CCH) {
    const float* wr = W_ts + (long)col * NTEMB;
    float acc = b_ts[col];
    for (int k = 0; k < NTEMB; k++) acc += sb[k] * wr[k];
    ss[b * 2 * CCH + col] = acc;
  } else {
    int c2 = col - 2 * CCH;
    const float* wr = W_tp + (long)c2 * NTEMB;
    float acc = b_tp[c2];
    for (int k = 0; k < NTEMB; k++) acc += sb[k] * wr[k];
    tp[b * CCH + c2] = acc;
  }
}

// h = x*(1+scale)+shift, stored transposed as bf16 hT[b][pixel][chan]
__global__ __launch_bounds__(256) void spc_affine_T(const float* __restrict__ x,
                                                    const float* __restrict__ ss,
                                                    unsigned short* __restrict__ hT) {
  __shared__ unsigned short tile[64][66];
  int b = blockIdx.z;
  int c0 = blockIdx.y * 64;
  int p0 = blockIdx.x * 64;
  int t = threadIdx.x;
  int pl = t & 63, cr = t >> 6;
  for (int r = 0; r < 16; r++) {
    int cl = r * 4 + cr;
    int c = c0 + cl;
    float sc = 1.0f + ss[b * 2 * CCH + c];
    float sh = ss[b * 2 * CCH + CCH + c];
    float v = x[((long)(b * CCH + c)) * HW + p0 + pl];
    tile[cl][pl] = f32_to_bf16(v * sc + sh);
  }
  __syncthreads();
  int cl2 = t & 63, pr = t >> 6;
  for (int r = 0; r < 16; r++) {
    int pl2 = r * 4 + pr;
    hT[((long)(b * HW + p0 + pl2)) * CCH + c0 + cl2] = tile[cl2][pl2];
  }
}

// p = concat(gap, tp), gap = (1+scale)*xmean + shift
__global__ void spc_build_p(const float* __restrict__ ss, const float* __restrict__ xmean,
                            const float* __restrict__ tp, float* __restrict__ p) {
  int i = blockIdx.x * 256 + threadIdx.x;
  if (i >= NB * 2 * CCH) return;
  int b = i / (2 * CCH);
  int k = i % (2 * CCH);
  float v;
  if (k < CCH)
    v = (1.0f + ss[b * 2 * CCH + k]) * xmean[b * CCH + k] + ss[b * 2 * CCH + CCH + k];
  else
    v = tp[b * CCH + (k - CCH)];
  p[i] = v;
}

// hid = silu(p @ W_m1^T + b_m1)
__global__ void spc_mlp1(const float* __restrict__ p, const float* __restrict__ W_m1,
                         const float* __restrict__ b_m1, float* __restrict__ hid) {
  int i = blockIdx.x * 256 + threadIdx.x;
  if (i >= NB * NHID) return;
  int b = i / NHID, o = i % NHID;
  const float* pb = p + b * 2 * CCH;
  const float* wr = W_m1 + (long)o * 2 * CCH;
  float acc = b_m1[o];
  for (int k = 0; k < 2 * CCH; k++) acc += pb[k] * wr[k];
  hid[i] = silu_f(acc);
}

// prompt = hid @ W_m2^T + b_m2
__global__ void spc_mlp2(const float* __restrict__ hid, const float* __restrict__ W_m2,
                         const float* __restrict__ b_m2, float* __restrict__ prompt) {
  int i = blockIdx.x * 256 + threadIdx.x;
  if (i >= NB * CCH) return;
  int b = i / CCH, o = i % CCH;
  const float* hb = hid + b * NHID;
  const float* wr = W_m2 + (long)o * NHID;
  float acc = b_m2[o];
  for (int k = 0; k < NHID; k++) acc += hb[k] * wr[k];
  prompt[i] = acc;
}

// exact top-K set via rank counting (ties: lower index wins); order-free compaction
__global__ __launch_bounds__(256) void spc_topk(const float* __restrict__ prompt,
                                                int* __restrict__ idx) {
  __shared__ float pr[CCH];
  __shared__ int cnt;
  int b = blockIdx.x, t = threadIdx.x;
  for (int i = t; i < CCH; i += 256) pr[i] = prompt[b * CCH + i];
  if (t == 0) cnt = 0;
  __syncthreads();
  for (int i = t; i < CCH; i += 256) {
    float v = pr[i];
    int rank = 0;
    for (int j = 0; j < CCH; j++) {
      float w = pr[j];
      rank += (w > v) || (w == v && j < i);
    }
    if (rank < KSEL) {
      int pos = atomicAdd(&cnt, 1);
      idx[b * KSEL + pos] = i;
    }
  }
}

// Wsel[b][j][c] = W[idx[b][j]][c] (bf16), bsel[b][j] = bias[idx[b][j]]
__global__ void spc_gather_w(const float* __restrict__ W, const float* __restrict__ bvec,
                             const int* __restrict__ idx, unsigned short* __restrict__ Wsel,
                             float* __restrict__ bsel) {
  long i = (long)blockIdx.x * 256 + threadIdx.x;
  if (i >= (long)NB * KSEL * CCH) return;
  int c = (int)(i % CCH);
  long tjb = i / CCH;
  int j = (int)(tjb % KSEL);
  int b = (int)(tjb / KSEL);
  int ch = idx[b * KSEL + j];
  Wsel[i] = f32_to_bf16(W[(long)ch * CCH + c]);
  if (c == 0) bsel[b * KSEL + j] = bvec[ch];
}

// Wo_sel[b][o][j] = Wo[o][idx[b][j]] (bf16)
__global__ void spc_gather_wo(const float* __restrict__ Wo, const int* __restrict__ idx,
                              unsigned short* __restrict__ Wos) {
  long i = (long)blockIdx.x * 256 + threadIdx.x;
  if (i >= (long)NB * CCH * KSEL) return;
  int j = (int)(i % KSEL);
  long tob = i / KSEL;
  int o = (int)(tob % CCH);
  int b = (int)(tob / CCH);
  Wos[i] = f32_to_bf16(Wo[(long)o * CCH + idx[b * KSEL + j]]);
}

// ---------------------------------------------------------------------------
// Generic NT GEMM: C[m][n] = alpha * sum_k A[m][k]*Bt[n][k] (+ bias[m])
// bf16 in, f32 accumulate via v_wmma_f32_16x16x32_bf16. 128x128 tile, BK=32.
// Ping-pong LDS buffers filled by GLOBAL_LOAD_ASYNC_TO_LDS_B128 (ASYNCcnt),
// overlapping next-tile DMA with current-tile WMMA.
#define LDK 40  // padded LDS row stride (halves); 80B keeps 16B alignment

template <bool OUT_BF16, bool OUT_TRANS, bool HAS_BIAS>
__global__ __launch_bounds__(256) void spc_gemm_nt(
    const unsigned short* __restrict__ A, const unsigned short* __restrict__ Bt,
    void* __restrict__ Out, const float* __restrict__ bias, int K, long sA, long sB,
    long sO, int sBias, float alpha, int ldo) {
  __shared__ unsigned short As[2][128 * LDK];
  __shared__ unsigned short Bs[2][128 * LDK];
  int b = blockIdx.z;
  const unsigned short* Ab = A + (long)b * sA;
  const unsigned short* Bb = Bt + (long)b * sB;
  int m0 = blockIdx.y * 128;
  int n0 = blockIdx.x * 128;
  int tid = threadIdx.x;
  int lane = tid & 31;
  int wave = tid >> 5;
  int wm = wave & 3;   // 4 waves along M (32 rows each)
  int wn = wave >> 2;  // 2 waves along N (64 cols each)
  int lm = lane & 15;
  int lh = lane >> 4;

  // per-thread tile slots (128x32 halves = 512 uint4; 2 per thread per matrix)
  int row0 = tid >> 2, kq0 = (tid & 3) * 8;
  int row1 = (tid + 256) >> 2, kq1 = ((tid + 256) & 3) * 8;

  floatx8 acc[2][4];
#pragma unroll
  for (int i = 0; i < 2; i++)
#pragma unroll
    for (int j = 0; j < 4; j++)
#pragma unroll
      for (int e = 0; e < 8; e++) acc[i][j][e] = 0.0f;

  int nk = K >> 5;

#if SPC_ASYNC
  unsigned asb[2] = {(unsigned)(uintptr_t)&As[0][0], (unsigned)(uintptr_t)&As[1][0]};
  unsigned bsb[2] = {(unsigned)(uintptr_t)&Bs[0][0], (unsigned)(uintptr_t)&Bs[1][0]};
  auto issue = [&](int bufi, int k0) {
    __builtin_amdgcn_global_load_async_to_lds_b128(
        (gas_v4i*)(uintptr_t)(Ab + (long)(m0 + row0) * K + k0 + kq0),
        (lds_v4i*)(asb[bufi] + (unsigned)(row0 * LDK + kq0) * 2u), 0, 0);
    __builtin_amdgcn_global_load_async_to_lds_b128(
        (gas_v4i*)(uintptr_t)(Ab + (long)(m0 + row1) * K + k0 + kq1),
        (lds_v4i*)(asb[bufi] + (unsigned)(row1 * LDK + kq1) * 2u), 0, 0);
    __builtin_amdgcn_global_load_async_to_lds_b128(
        (gas_v4i*)(uintptr_t)(Bb + (long)(n0 + row0) * K + k0 + kq0),
        (lds_v4i*)(bsb[bufi] + (unsigned)(row0 * LDK + kq0) * 2u), 0, 0);
    __builtin_amdgcn_global_load_async_to_lds_b128(
        (gas_v4i*)(uintptr_t)(Bb + (long)(n0 + row1) * K + k0 + kq1),
        (lds_v4i*)(bsb[bufi] + (unsigned)(row1 * LDK + kq1) * 2u), 0, 0);
  };
  issue(0, 0);
#endif

  for (int it = 0; it < nk; ++it) {
    int k0 = it << 5;
#if SPC_ASYNC
    int cur = it & 1;
    __builtin_amdgcn_s_wait_asynccnt(0);  // tile `it` DMA complete (this wave)
    __syncthreads();                      // visible to all waves
    if (it + 1 < nk) issue(cur ^ 1, k0 + 32);  // overlap next DMA with WMMA
    const unsigned short* AsC = As[cur];
    const unsigned short* BsC = Bs[cur];
#else
    *((uint4*)(As[0] + row0 * LDK + kq0)) =
        *((const uint4*)(Ab + (long)(m0 + row0) * K + k0 + kq0));
    *((uint4*)(As[0] + row1 * LDK + kq1)) =
        *((const uint4*)(Ab + (long)(m0 + row1) * K + k0 + kq1));
    *((uint4*)(Bs[0] + row0 * LDK + kq0)) =
        *((const uint4*)(Bb + (long)(n0 + row0) * K + k0 + kq0));
    *((uint4*)(Bs[0] + row1 * LDK + kq1)) =
        *((const uint4*)(Bb + (long)(n0 + row1) * K + k0 + kq1));
    __syncthreads();
    const unsigned short* AsC = As[0];
    const unsigned short* BsC = Bs[0];
#endif

    bf16x16 af[2], bfr[4];
#pragma unroll
    for (int mi = 0; mi < 2; mi++) {
      const unsigned short* base = AsC + (wm * 32 + mi * 16 + lm) * LDK;
      union { uint4 u[2]; bf16x16 v; } cvt;
      cvt.u[0] = *((const uint4*)(base + lh * 8));       // k = h*8 .. h*8+7
      cvt.u[1] = *((const uint4*)(base + 16 + lh * 8));  // k = 16+h*8 ..
      af[mi] = cvt.v;
    }
#pragma unroll
    for (int ni = 0; ni < 4; ni++) {
      const unsigned short* base = BsC + (wn * 64 + ni * 16 + lm) * LDK;
      union { uint4 u[2]; bf16x16 v; } cvt;
      cvt.u[0] = *((const uint4*)(base + lh * 8));       // k = h*8 .. h*8+7
      cvt.u[1] = *((const uint4*)(base + 16 + lh * 8));  // k = 16+h*8 ..
      bfr[ni] = cvt.v;
    }
#pragma unroll
    for (int mi = 0; mi < 2; mi++)
#pragma unroll
      for (int ni = 0; ni < 4; ni++)
        acc[mi][ni] = __builtin_amdgcn_wmma_f32_16x16x32_bf16(
            false, af[mi], false, bfr[ni], (short)0, acc[mi][ni], false, false);
    __syncthreads();
  }

  int mbase = m0 + wm * 32;
  int nbase = n0 + wn * 64;
#pragma unroll
  for (int mi = 0; mi < 2; mi++) {
#pragma unroll
    for (int ni = 0; ni < 4; ni++) {
      int n = nbase + ni * 16 + lm;
#pragma unroll
      for (int j = 0; j < 8; j++) {
        int m = mbase + mi * 16 + lh * 8 + j;  // D: m = j + 8*(lane/16)
        float v = acc[mi][ni][j] * alpha;
        if (HAS_BIAS) v += bias[b * sBias + m];
        long o = OUT_TRANS ? ((long)b * sO + (long)n * ldo + m)
                           : ((long)b * sO + (long)m * ldo + n);
        if (OUT_BF16) ((unsigned short*)Out)[o] = f32_to_bf16(v);
        else ((float*)Out)[o] = v;
      }
    }
  }
}

// row softmax of S[b][r][0..383] -> bf16 P
__global__ __launch_bounds__(256) void spc_softmax(const float* __restrict__ S,
                                                   unsigned short* __restrict__ P) {
  __shared__ float red[256];
  int b = blockIdx.y, r = blockIdx.x, t = threadIdx.x;
  const float* row = S + ((long)(b * KSEL + r)) * KSEL;
  unsigned short* prow = P + ((long)(b * KSEL + r)) * KSEL;
  float m = -3.0e38f;
  for (int i = t; i < KSEL; i += 256) m = fmaxf(m, row[i]);
  red[t] = m; __syncthreads();
  for (int st = 128; st > 0; st >>= 1) {
    if (t < st) red[t] = fmaxf(red[t], red[t + st]);
    __syncthreads();
  }
  m = red[0]; __syncthreads();
  float s = 0.0f;
  for (int i = t; i < KSEL; i += 256) s += expf(row[i] - m);
  red[t] = s; __syncthreads();
  for (int st = 128; st > 0; st >>= 1) {
    if (t < st) red[t] += red[t + st];
    __syncthreads();
  }
  float inv = 1.0f / red[0];
  for (int i = t; i < KSEL; i += 256) prow[i] = f32_to_bf16(expf(row[i] - m) * inv);
}

// ---------------------------------------------------------------------------
extern "C" void kernel_launch(void* const* d_in, const int* in_sizes, int n_in,
                              void* d_out, int out_size, void* d_ws, size_t ws_size,
                              hipStream_t stream) {
  const float* x    = (const float*)d_in[0];
  const float* temb = (const float*)d_in[1];
  const float* W_ts = (const float*)d_in[2];
  const float* b_ts = (const float*)d_in[3];
  const float* W_tp = (const float*)d_in[4];
  const float* b_tp = (const float*)d_in[5];
  const float* W_m1 = (const float*)d_in[6];
  const float* b_m1 = (const float*)d_in[7];
  const float* W_m2 = (const float*)d_in[8];
  const float* b_m2 = (const float*)d_in[9];
  const float* Wq = (const float*)d_in[10];
  const float* bq = (const float*)d_in[11];
  const float* Wk = (const float*)d_in[12];
  const float* bk = (const float*)d_in[13];
  const float* Wv = (const float*)d_in[14];
  const float* bv = (const float*)d_in[15];
  const float* Wo = (const float*)d_in[16];
  const float* bo = (const float*)d_in[17];
  float* out = (float*)d_out;

  char* wsb = (char*)d_ws;
  size_t off = 0;
  auto alloc = [&](size_t bytes) -> void* {
    void* p = wsb + off;
    off = (off + bytes + 255) & ~(size_t)255;
    return p;
  };
  float* st     = (float*)alloc((size_t)NB * NTEMB * 4);
  float* ssb    = (float*)alloc((size_t)NB * 2 * CCH * 4);
  float* tpb    = (float*)alloc((size_t)NB * CCH * 4);
  float* xmeanb = (float*)alloc((size_t)NB * CCH * 4);
  float* pbuf   = (float*)alloc((size_t)NB * 2 * CCH * 4);
  float* hid    = (float*)alloc((size_t)NB * NHID * 4);
  float* prompt = (float*)alloc((size_t)NB * CCH * 4);
  int*   idx    = (int*)alloc((size_t)NB * KSEL * 4);
  float* bq_sel = (float*)alloc((size_t)NB * KSEL * 4);
  float* bk_sel = (float*)alloc((size_t)NB * KSEL * 4);
  float* bv_sel = (float*)alloc((size_t)NB * KSEL * 4);
  unsigned short* hT     = (unsigned short*)alloc((size_t)NB * HW * CCH * 2);
  unsigned short* Wq_sel = (unsigned short*)alloc((size_t)NB * KSEL * CCH * 2);
  unsigned short* Wk_sel = (unsigned short*)alloc((size_t)NB * KSEL * CCH * 2);
  unsigned short* Wv_sel = (unsigned short*)alloc((size_t)NB * KSEL * CCH * 2);
  unsigned short* Wo_sel = (unsigned short*)alloc((size_t)NB * CCH * KSEL * 2);
  unsigned short* qb     = (unsigned short*)alloc((size_t)NB * KSEL * HW * 2);
  unsigned short* kb     = (unsigned short*)alloc((size_t)NB * KSEL * HW * 2);
  unsigned short* vT     = (unsigned short*)alloc((size_t)NB * HW * KSEL * 2);
  float*          Sb     = (float*)alloc((size_t)NB * KSEL * KSEL * 4);
  unsigned short* Pb     = (unsigned short*)alloc((size_t)NB * KSEL * KSEL * 2);
  unsigned short* Ot     = (unsigned short*)alloc((size_t)NB * HW * KSEL * 2);

  // 1) small dense path
  spc_xmean<<<dim3(CCH, NB), 256, 0, stream>>>(x, xmeanb);
  spc_silu_temb<<<(NB * NTEMB + 255) / 256, 256, 0, stream>>>(temb, st);
  spc_time_proj<<<(NB * 3 * CCH + 255) / 256, 256, 0, stream>>>(st, W_ts, b_ts, W_tp, b_tp,
                                                                ssb, tpb);
  // 2) h (affine), transposed bf16 staging
  spc_affine_T<<<dim3(HW / 64, CCH / 64, NB), 256, 0, stream>>>(x, ssb, hT);
  // 3) prompt MLP + top-k
  spc_build_p<<<(NB * 2 * CCH + 255) / 256, 256, 0, stream>>>(ssb, xmeanb, tpb, pbuf);
  spc_mlp1<<<(NB * NHID + 255) / 256, 256, 0, stream>>>(pbuf, W_m1, b_m1, hid);
  spc_mlp2<<<(NB * CCH + 255) / 256, 256, 0, stream>>>(hid, W_m2, b_m2, prompt);
  spc_topk<<<NB, 256, 0, stream>>>(prompt, idx);
  // 4) gather selected weights/biases (bf16)
  long gw = (long)NB * KSEL * CCH;
  spc_gather_w<<<(unsigned)((gw + 255) / 256), 256, 0, stream>>>(Wq, bq, idx, Wq_sel, bq_sel);
  spc_gather_w<<<(unsigned)((gw + 255) / 256), 256, 0, stream>>>(Wk, bk, idx, Wk_sel, bk_sel);
  spc_gather_w<<<(unsigned)((gw + 255) / 256), 256, 0, stream>>>(Wv, bv, idx, Wv_sel, bv_sel);
  spc_gather_wo<<<(unsigned)((gw + 255) / 256), 256, 0, stream>>>(Wo, idx, Wo_sel);

  // 5) q/k/v projections: [384x768] x [768x4096]  (v stored transposed)
  dim3 g1(HW / 128, KSEL / 128, NB);
  spc_gemm_nt<true, false, true><<<g1, 256, 0, stream>>>(
      Wq_sel, hT, qb, bq_sel, CCH, (long)KSEL * CCH, (long)HW * CCH, (long)KSEL * HW,
      KSEL, 1.0f, HW);
  spc_gemm_nt<true, false, true><<<g1, 256, 0, stream>>>(
      Wk_sel, hT, kb, bk_sel, CCH, (long)KSEL * CCH, (long)HW * CCH, (long)KSEL * HW,
      KSEL, 1.0f, HW);
  spc_gemm_nt<true, true, true><<<g1, 256, 0, stream>>>(
      Wv_sel, hT, vT, bv_sel, CCH, (long)KSEL * CCH, (long)HW * CCH, (long)HW * KSEL,
      KSEL, 1.0f, KSEL);

  // 6) scores S = q k^T * K^-0.5 : [384x4096] x [384x4096]^T
  dim3 g2(KSEL / 128, KSEL / 128, NB);
  spc_gemm_nt<false, false, false><<<g2, 256, 0, stream>>>(
      qb, kb, Sb, (const float*)nullptr, HW, (long)KSEL * HW, (long)KSEL * HW,
      (long)KSEL * KSEL, 0, rsqrtf((float)KSEL), KSEL);
  spc_softmax<<<dim3(KSEL, NB), 256, 0, stream>>>(Sb, Pb);

  // 7) O = P v  (stored transposed): [384x384] x [4096x384]^T
  dim3 g3(HW / 128, KSEL / 128, NB);
  spc_gemm_nt<true, true, false><<<g3, 256, 0, stream>>>(
      Pb, vT, Ot, (const float*)nullptr, KSEL, (long)KSEL * KSEL, (long)HW * KSEL,
      (long)HW * KSEL, 0, 1.0f, KSEL);

  // 8) Y = Wo_sel O + bo -> d_out : [768x384] x [4096x384]^T
  dim3 g4(HW / 128, CCH / 128, NB);
  spc_gemm_nt<false, false, true><<<g4, 256, 0, stream>>>(
      Wo_sel, Ot, out, bo, KSEL, (long)CCH * KSEL, (long)HW * KSEL, (long)CCH * HW,
      0, 1.0f, HW);

  (void)in_sizes; (void)n_in; (void)out_size; (void)ws_size;
}
